// Model_45664092291553
// MI455X (gfx1250) — matrix operand order repaired and checked by reading, not compiled, not run on hardware
//
#include <hip/hip_runtime.h>
#include <hip/hip_bf16.h>

typedef __attribute__((ext_vector_type(16))) __bf16 v16bf;
typedef __attribute__((ext_vector_type(8)))  __bf16 v8bf;
typedef __attribute__((ext_vector_type(4)))  __bf16 v4bf;
typedef __attribute__((ext_vector_type(8)))  float  v8f;

#define BATCH 8
#define SEQL  3072
#define CH    512
#define MROWS (BATCH * SEQL)   // 24576

// ---------------------------------------------------------------------------
// GEMM: C[M x N] = A[M x K](f32) * W[K x N], W pre-transposed to bf16 [N][K].
// WG = 256 thr = 8 waves (2x4), wave = 32x32 = 2x2 wmma tiles, WG tile 64x128.
// Double-buffered LDS; 128-bit staging loads/stores; bf16 WMMA, f32 accum.
// ---------------------------------------------------------------------------
__global__ __launch_bounds__(256) void gemm_bf16w(const float* __restrict__ A,
                                                  const __bf16* __restrict__ Wt,
                                                  float* __restrict__ C,
                                                  int M, int N, int K) {
  __shared__ __align__(16) __bf16 As[2][64][40];    // pad 8 -> rows 16B aligned
  __shared__ __align__(16) __bf16 Ws[2][128][40];   // n-major (k contiguous)
  const int tid  = threadIdx.x;
  const int lane = tid & 31;
  const int wv   = tid >> 5;
  const int wm   = wv >> 2, wn = wv & 3;
  const int m0   = blockIdx.x * 64, n0 = blockIdx.y * 128;
  const int h    = lane >> 4, ln = lane & 15;

  int rA[2], cA[2], nW[2], cW[2];
#pragma unroll
  for (int i = 0; i < 2; ++i) {
    int chunk = tid + 256 * i;
    rA[i] = chunk >> 3; cA[i] = (chunk & 7) * 4;
    nW[i] = chunk >> 2; cW[i] = (chunk & 3) * 8;
  }
  const float*  aBase[2] = { A + (size_t)(m0 + rA[0]) * K + cA[0],
                             A + (size_t)(m0 + rA[1]) * K + cA[1] };
  const __bf16* wBase[2] = { Wt + (size_t)(n0 + nW[0]) * K + cW[0],
                             Wt + (size_t)(n0 + nW[1]) * K + cW[1] };

  v8f acc[2][2] = {};
  float4 aReg[2];
  v8bf   wReg[2];

#pragma unroll
  for (int i = 0; i < 2; ++i) {
    aReg[i] = *(const float4*)(aBase[i]);
    wReg[i] = *(const v8bf*)(wBase[i]);
  }
#pragma unroll
  for (int i = 0; i < 2; ++i) {
    v4bf pk;
    pk[0] = (__bf16)aReg[i].x; pk[1] = (__bf16)aReg[i].y;
    pk[2] = (__bf16)aReg[i].z; pk[3] = (__bf16)aReg[i].w;
    *(v4bf*)(&As[0][rA[i]][cA[i]]) = pk;
    *(v8bf*)(&Ws[0][nW[i]][cW[i]]) = wReg[i];
  }
  __syncthreads();

  const int steps = K >> 5;                       // K / 32
  for (int s = 0; s < steps; ++s) {
    const int cur = s & 1, nxt = cur ^ 1;
    const int kk2 = (s + 1) << 5;

    if (s + 1 < steps) {                          // issue next tile's loads early
#pragma unroll
      for (int i = 0; i < 2; ++i) {
        aReg[i] = *(const float4*)(aBase[i] + kk2);
        wReg[i] = *(const v8bf*)(wBase[i] + kk2);
      }
      __builtin_prefetch(aBase[0] + kk2 + 32, 0, 3);
    }

    v16bf af[2], bw[2];
#pragma unroll
    for (int t = 0; t < 2; ++t) {
      const __bf16* row = &As[cur][wm * 32 + t * 16 + ln][0];
      v8bf lo = *(const v8bf*)(row + 16 * h);
      v8bf hi = *(const v8bf*)(row + 16 * h + 8);
      af[t] = __builtin_shufflevector(lo, hi, 0,1,2,3,4,5,6,7,8,9,10,11,12,13,14,15);
      const __bf16* col = &Ws[cur][wn * 32 + t * 16 + ln][0];
      bw[t] = *(const v16bf*)(col + 16 * h);
    }
#pragma unroll
    for (int tm = 0; tm < 2; ++tm)
#pragma unroll
      for (int tn = 0; tn < 2; ++tn)
        acc[tm][tn] = __builtin_amdgcn_wmma_f32_16x16x32_bf16(
            false, af[tm], false, bw[tn], (short)0, acc[tm][tn], false, false);

    if (s + 1 < steps) {
#pragma unroll
      for (int i = 0; i < 2; ++i) {
        v4bf pk;
        pk[0] = (__bf16)aReg[i].x; pk[1] = (__bf16)aReg[i].y;
        pk[2] = (__bf16)aReg[i].z; pk[3] = (__bf16)aReg[i].w;
        *(v4bf*)(&As[nxt][rA[i]][cA[i]]) = pk;
        *(v8bf*)(&Ws[nxt][nW[i]][cW[i]]) = wReg[i];
      }
    }
    __syncthreads();
  }

#pragma unroll
  for (int tm = 0; tm < 2; ++tm)
#pragma unroll
    for (int tn = 0; tn < 2; ++tn)
#pragma unroll
      for (int r = 0; r < 8; ++r) {
        int row = m0 + wm * 32 + tm * 16 + r + 8 * h;
        int col = n0 + wn * 32 + tn * 16 + ln;
        C[(size_t)row * N + col] = acc[tm][tn][r];
      }
}

// ---------------------------------------------------------------------------
// Weight transpose+convert: f32 [K=512][N=512] -> bf16 [N][K]
// ---------------------------------------------------------------------------
__global__ __launch_bounds__(256) void wtrans_bf16(const float* __restrict__ in,
                                                   __bf16* __restrict__ out) {
  __shared__ float tile[32][33];
  int n0 = blockIdx.x * 32, k0 = blockIdx.y * 32;
  int x = threadIdx.x, y0 = threadIdx.y;
#pragma unroll
  for (int i = 0; i < 4; ++i) {
    int y = y0 + i * 8;
    tile[y][x] = in[(size_t)(k0 + y) * CH + n0 + x];
  }
  __syncthreads();
#pragma unroll
  for (int i = 0; i < 4; ++i) {
    int y = y0 + i * 8;
    out[(size_t)(n0 + y) * CH + k0 + x] = (__bf16)tile[x][y];
  }
}

// ---------------------------------------------------------------------------
// Transpose [B][L][C] f32 -> [B][C][L] bf16 (channel-major for correlation)
// ---------------------------------------------------------------------------
__global__ __launch_bounds__(256) void transpose_cvt_bf16(const float* __restrict__ in,
                                                          __bf16* __restrict__ out) {
  __shared__ float tile[32][33];
  int b = blockIdx.z;
  int c0 = blockIdx.x * 32, l0 = blockIdx.y * 32;
  int x = threadIdx.x, y0 = threadIdx.y;
#pragma unroll
  for (int i = 0; i < 4; ++i) {
    int y = y0 + i * 8;
    tile[y][x] = in[((size_t)b * SEQL + l0 + y) * CH + c0 + x];
  }
  __syncthreads();
#pragma unroll
  for (int i = 0; i < 4; ++i) {
    int y = y0 + i * 8;
    out[((size_t)b * CH + c0 + y) * SEQL + l0 + x] = (__bf16)tile[x][y];
  }
}

// Transpose corr [B][C][L] f32 -> [B][L][C] f32 (into d_out corr segment)
__global__ __launch_bounds__(256) void transpose_f32(const float* __restrict__ in,
                                                     float* __restrict__ out) {
  __shared__ float tile[32][33];
  int b = blockIdx.z;
  int l0 = blockIdx.x * 32, c0 = blockIdx.y * 32;
  int x = threadIdx.x, y0 = threadIdx.y;
#pragma unroll
  for (int i = 0; i < 4; ++i) {
    int y = y0 + i * 8;
    tile[y][x] = in[((size_t)b * CH + c0 + y) * SEQL + l0 + x];
  }
  __syncthreads();
#pragma unroll
  for (int i = 0; i < 4; ++i) {
    int y = y0 + i * 8;
    out[((size_t)b * SEQL + l0 + y) * CH + c0 + x] = tile[x][y];
  }
}

// ---------------------------------------------------------------------------
// Per-channel circular autocorrelation as WMMA GEMM.
// corr[tau] = sum_t q[(t+tau)%L] * k[t];  tau = 16*tau1 + tau2.
// LDS fill uses GLOBAL_LOAD_ASYNC_TO_LDS_B128 (ASYNCcnt) — pure aligned copy.
// ---------------------------------------------------------------------------
__global__ __launch_bounds__(128) void corr_wmma(const __bf16* __restrict__ qT,
                                                 const __bf16* __restrict__ kT,
                                                 float* __restrict__ corrWS,
                                                 float* __restrict__ meanWS) {
  __shared__ __align__(16) __bf16 qd[2 * SEQL];      // q duplicated: idx in [0,2L)
  __shared__ __align__(16) __bf16 kd[SEQL + 32];     // kd[i] = k[(i-16+L)%L]
  const int blk = blockIdx.x;
  const int b = blk >> 9, c = blk & 511;
  const int tid = threadIdx.x, lane = tid & 31, wv = tid >> 5;
  const size_t chan = ((size_t)b * CH + c) * SEQL;

  // async global->LDS fill: each lane moves 16B per op, all 16B-aligned
  for (int i = tid * 8; i < SEQL; i += 128 * 8) {
    unsigned qo0 = (unsigned)(uintptr_t)&qd[i];
    unsigned qo1 = (unsigned)(uintptr_t)&qd[i + SEQL];
    unsigned ko  = (unsigned)(uintptr_t)&kd[i + 16];
    const __bf16* gq = &qT[chan + i];
    const __bf16* gk = &kT[chan + i];
    asm volatile("global_load_async_to_lds_b128 %0, %1, off"
                 :: "v"(qo0), "v"(gq) : "memory");
    asm volatile("global_load_async_to_lds_b128 %0, %1, off"
                 :: "v"(qo1), "v"(gq) : "memory");
    asm volatile("global_load_async_to_lds_b128 %0, %1, off"
                 :: "v"(ko), "v"(gk) : "memory");
  }
  if (tid < 16) kd[tid] = kT[chan + SEQL - 16 + tid];   // wrap tail (DS path)
  asm volatile("s_wait_asynccnt 0" ::: "memory");
  __syncthreads();

  const int h = lane >> 4, ln = lane & 15;
  v8f acc[3] = {};

  for (int t0 = 0; t0 < SEQL; t0 += 32) {
    // B fragment: element j -> K = j + 16h; B[t,n] = kd[t - n + 16]
    v16bf bfr;
    {
      int boff = t0 + 16 * h - ln + 16;
#pragma unroll
      for (int j = 0; j < 16; ++j) bfr[j] = kd[boff + j];
    }
#pragma unroll
    for (int i = 0; i < 3; ++i) {
      int tau1 = (wv * 3 + i) * 16 + ln;             // A row = lag/16
      const __bf16* row = &qd[t0 + 16 * tau1];       // 16B-aligned
      v8bf lo = *(const v8bf*)(row + 16 * h);
      v8bf hi = *(const v8bf*)(row + 16 * h + 8);
      v16bf af = __builtin_shufflevector(lo, hi, 0,1,2,3,4,5,6,7,8,9,10,11,12,13,14,15);
      acc[i] = __builtin_amdgcn_wmma_f32_16x16x32_bf16(
          false, af, false, bfr, (short)0, acc[i], false, false);
    }
  }

#pragma unroll
  for (int i = 0; i < 3; ++i) {
    int tileId = wv * 3 + i;
#pragma unroll
    for (int r = 0; r < 8; ++r) {
      int tau = tileId * 256 + 16 * (r + 8 * h) + ln;
      float val = acc[i][r];
      corrWS[chan + tau] = val;
      unsafeAtomicAdd(&meanWS[b * SEQL + tau], val * (1.0f / (float)CH));
    }
  }
}

// ---------------------------------------------------------------------------
// Top-8 lags of the channel-mean + softmax (one WG per batch element)
// ---------------------------------------------------------------------------
__global__ __launch_bounds__(256) void topk_softmax(const float* __restrict__ meanWS,
                                                    float* __restrict__ topw,
                                                    int* __restrict__ topd) {
  __shared__ float sm[SEQL];
  __shared__ float rv[256];
  __shared__ int   ri[256];
  __shared__ float wvv[8];
  __shared__ int   wii[8];
  int b = blockIdx.x, tid = threadIdx.x;
  for (int i = tid; i < SEQL; i += 256) sm[i] = meanWS[b * SEQL + i];
  __syncthreads();
  for (int it = 0; it < 8; ++it) {
    float best = -3.0e38f; int bi = 0;
    for (int i = tid; i < SEQL; i += 256) {
      float v = sm[i];
      if (v > best) { best = v; bi = i; }
    }
    rv[tid] = best; ri[tid] = bi;
    __syncthreads();
    for (int s = 128; s > 0; s >>= 1) {
      if (tid < s && rv[tid + s] > rv[tid]) { rv[tid] = rv[tid + s]; ri[tid] = ri[tid + s]; }
      __syncthreads();
    }
    if (tid == 0) { wvv[it] = rv[0]; wii[it] = ri[0]; sm[ri[0]] = -3.0e38f; }
    __syncthreads();
  }
  if (tid == 0) {
    float mx = wvv[0];
    for (int j = 1; j < 8; ++j) mx = fmaxf(mx, wvv[j]);
    float s = 0.f, e[8];
    for (int j = 0; j < 8; ++j) { e[j] = __expf(wvv[j] - mx); s += e[j]; }
    for (int j = 0; j < 8; ++j) { topw[b * 8 + j] = e[j] / s; topd[b * 8 + j] = wii[j]; }
  }
}

// ---------------------------------------------------------------------------
// V[b,l,c] = sum_j w[b,j] * v[b, (l + delay_j) % L, c]   (float4 per thread)
// ---------------------------------------------------------------------------
__global__ __launch_bounds__(256) void aggregate_kernel(const float* __restrict__ v,
                                                        const float* __restrict__ topw,
                                                        const int* __restrict__ topd,
                                                        float* __restrict__ V) {
  __shared__ float ws8[8];
  __shared__ int   ds8[8];
  int idx4 = blockIdx.x * 256 + threadIdx.x;      // over NQ/4; b uniform per block
  int b = idx4 / (SEQL * CH / 4);
  if (threadIdx.x < 8) { ws8[threadIdx.x] = topw[b * 8 + threadIdx.x];
                         ds8[threadIdx.x] = topd[b * 8 + threadIdx.x]; }
  __syncthreads();
  int rem = idx4 - b * (SEQL * CH / 4);
  int l = rem / (CH / 4);
  int c4 = (rem & (CH / 4 - 1)) * 4;
  float4 acc = make_float4(0.f, 0.f, 0.f, 0.f);
#pragma unroll
  for (int j = 0; j < 8; ++j) {
    int ll = l + ds8[j];
    if (ll >= SEQL) ll -= SEQL;
    const float4 t = *(const float4*)(&v[((size_t)b * SEQL + ll) * CH + c4]);
    float w = ws8[j];
    acc.x += w * t.x; acc.y += w * t.y; acc.z += w * t.z; acc.w += w * t.w;
  }
  *(float4*)(&V[(size_t)idx4 * 4]) = acc;
}

// ---------------------------------------------------------------------------
extern "C" void kernel_launch(void* const* d_in, const int* in_sizes, int n_in,
                              void* d_out, int out_size, void* d_ws, size_t ws_size,
                              hipStream_t stream) {
  const float* query = (const float*)d_in[0];
  const float* key   = (const float*)d_in[1];
  const float* value = (const float*)d_in[2];
  const float* Wq    = (const float*)d_in[3];
  const float* Wk    = (const float*)d_in[4];
  const float* Wv    = (const float*)d_in[5];
  const float* Wo    = (const float*)d_in[6];
  float* out = (float*)d_out;                         // [B,L,DM]
  float* corr_out = out + (size_t)MROWS * CH;         // [B,L,H,E]

  const size_t NQ = (size_t)MROWS * CH;               // 12582912
  float*  q      = (float*)d_ws;
  float*  k      = q + NQ;
  float*  v      = k + NQ;
  float*  Vb     = v + NQ;
  float*  corrWS = Vb + NQ;
  __bf16* qT     = (__bf16*)(corrWS + NQ);
  __bf16* kT     = qT + NQ;
  float*  meanWS = (float*)(kT + NQ);
  float*  topw   = meanWS + BATCH * SEQL;
  int*    topd   = (int*)(topw + BATCH * 8);
  __bf16* WqT    = (__bf16*)(topd + BATCH * 8);
  __bf16* WkT    = WqT + CH * CH;
  __bf16* WvT    = WkT + CH * CH;
  __bf16* WoT    = WvT + CH * CH;

  hipMemsetAsync(meanWS, 0, BATCH * SEQL * sizeof(float), stream);

  dim3 gW(CH / 32, CH / 32);
  wtrans_bf16<<<gW, dim3(32, 8), 0, stream>>>(Wq, WqT);
  wtrans_bf16<<<gW, dim3(32, 8), 0, stream>>>(Wk, WkT);
  wtrans_bf16<<<gW, dim3(32, 8), 0, stream>>>(Wv, WvT);
  wtrans_bf16<<<gW, dim3(32, 8), 0, stream>>>(Wo, WoT);

  dim3 gGemm(MROWS / 64, CH / 128);
  gemm_bf16w<<<gGemm, 256, 0, stream>>>(query, WqT, q, MROWS, CH, CH);
  gemm_bf16w<<<gGemm, 256, 0, stream>>>(key,   WkT, k, MROWS, CH, CH);
  gemm_bf16w<<<gGemm, 256, 0, stream>>>(value, WvT, v, MROWS, CH, CH);

  dim3 gT1(CH / 32, SEQL / 32, BATCH);
  transpose_cvt_bf16<<<gT1, dim3(32, 8), 0, stream>>>(q, qT);
  transpose_cvt_bf16<<<gT1, dim3(32, 8), 0, stream>>>(k, kT);

  corr_wmma<<<BATCH * CH, 128, 0, stream>>>(qT, kT, corrWS, meanWS);

  dim3 gT2(SEQL / 32, CH / 32, BATCH);
  transpose_f32<<<gT2, dim3(32, 8), 0, stream>>>(corrWS, corr_out);

  topk_softmax<<<BATCH, 256, 0, stream>>>(meanWS, topw, topd);

  aggregate_kernel<<<(unsigned)(NQ / 4 / 256), 256, 0, stream>>>(v, topw, topd, Vb);

  gemm_bf16w<<<gGemm, 256, 0, stream>>>(Vb, WoT, out, MROWS, CH, CH);
}
